// conv_block_75960791597956
// MI455X (gfx1250) — matrix operand order repaired and checked by reading, not compiled, not run on hardware
//
#include <hip/hip_runtime.h>
#include <hip/hip_bf16.h>

typedef __attribute__((ext_vector_type(16))) __bf16 v16bf;
typedef __attribute__((ext_vector_type(8)))  float  v8f;

constexpr int ICH1 = 16;   // input channels
constexpr int OCH  = 32;   // output channels per gate
constexpr int D    = 31;   // bands
constexpr int H    = 192;
constexpr int W    = 192;
constexpr int HW   = H * W;
constexpr int DHW  = D * HW;

// 16-bit A-matrix (16x32) element->K mapping per CDNA5 ISA 7.12.2:
// lanes 0-15: VGPR0..3 hold K=0..7, VGPR4..7 hold K=16..23
// lanes 16-31: K=8..15 and K=24..31
__device__ __forceinline__ int koffA(int j, int half) {
    return (j < 8) ? (half * 8 + j) : (16 + half * 8 + (j - 8));
}

// ---------------------------------------------------------------- fp32 -> bf16
__global__ void cvt_f32_bf16(const float* __restrict__ src, __bf16* __restrict__ dst, int n) {
    int i = blockIdx.x * blockDim.x + threadIdx.x;
    if (i < n) dst[i] = (__bf16)src[i];
}

// -------------------------------------------------- pack weights to B-fragments
// B (KxN, 16-bit) k order chosen here, same as the A staging: lane L -> N = L%16, half = L/16; elem j -> K = half*16 + j
// dst layout: [ntile(4)][kstep(KST)][lane(32)][elem(16)]
template<int ICH, int KST>
__global__ void pack_w(const float* __restrict__ w, __bf16* __restrict__ dst) {
    const int total = 4 * KST * 512;
    for (int idx = blockIdx.x * blockDim.x + threadIdx.x; idx < total;
         idx += gridDim.x * blockDim.x) {
        int ntile = idx / (KST * 512);
        int r     = idx % (KST * 512);
        int kstep = r / 512, q = r % 512, lane = q / 16, j = q % 16;
        int nloc = lane % 16, half = lane / 16;
        int K = kstep * 32 + half * 16 + j;
        float v = 0.0f;
        if (K < ICH * 27) v = w[(ntile * 16 + nloc) * ICH * 27 + K]; // K = ci*27 + tap
        dst[idx] = (__bf16)v;
    }
}

// -------------------------------------------- implicit-GEMM conv3d 3x3x3 + gates
// Block: 128 threads (4 waves). Tile: 16 pixels (one d,h row, 16 w) x 64 gate ch.
// Wave n computes N-tile n (16 channels). K = ICH*27 padded to KST*32.
template<int ICH, int KST>
__global__ __launch_bounds__(128)
void conv_qrnn_wmma(const __bf16* __restrict__ xin,    // (ICH, D, H, W)
                    const __bf16* __restrict__ wp,     // packed B fragments
                    float* __restrict__ zbuf,          // (OCH, D, H, W) relu gate
                    float* __restrict__ fbuf) {        // (OCH, D, H, W) sigmoid gate
    __shared__ __align__(32) __bf16 Afrag[KST * 512];  // [kstep][lane][16] fragment order

    const int bi  = blockIdx.x;
    const int d   = bi / (H * (W / 16));
    const int rem = bi % (H * (W / 16));
    const int h   = rem / (W / 16);
    const int w0  = (rem % (W / 16)) * 16;
    const int tid = threadIdx.x;

    // Stage im2col patch directly in A-fragment order (uniform loop, EXEC all-ones)
    for (int idx = tid; idx < KST * 512; idx += 128) {
        int kstep = idx >> 9;
        int q     = idx & 511;
        int lane  = q >> 4, j = q & 15;
        int M = lane & 15, half = lane >> 4;
        int k = kstep * 32 + half * 16 + j;
        __bf16 v = (__bf16)0.0f;
        if (k < ICH * 27) {
            int ci = k / 27, tap = k % 27;
            int kd = tap / 9, kh = (tap % 9) / 3, kw = tap % 3;
            int dd = d + kd - 1, hh = h + kh - 1, ww = w0 + M + kw - 1;
            if ((unsigned)dd < (unsigned)D && (unsigned)hh < (unsigned)H &&
                (unsigned)ww < (unsigned)W)
                v = xin[((size_t)(ci * D + dd) * H + hh) * W + ww];
        }
        Afrag[idx] = v;
    }
    __syncthreads();

    const int wave = tid >> 5;
    const int lane = tid & 31;
    v8f acc = {0.f, 0.f, 0.f, 0.f, 0.f, 0.f, 0.f, 0.f};
    const __bf16* wpn = wp + (size_t)wave * KST * 512 + lane * 16;

#pragma unroll 4
    for (int kstep = 0; kstep < KST; ++kstep) {
        v16bf a = *(const v16bf*)&Afrag[kstep * 512 + lane * 16];
        v16bf b = *(const v16bf*)(wpn + (size_t)kstep * 512);
        acc = __builtin_amdgcn_wmma_f32_16x16x32_bf16(
            /*neg_a=*/false, a, /*neg_b=*/false, b,
            /*c_mod=*/(short)0, acc, /*reuse_a=*/false, /*reuse_b=*/false);
    }

    // C/D layout: lane L, VGPR r -> M = r + 8*(L/16), N = L%16
    const int  nloc  = lane & 15;
    const int  mhalf = (lane >> 4) * 8;
    const bool is_f  = (wave >= 2);                       // gates[32:64] = forget
    const int  c     = (is_f ? (wave - 2) : wave) * 16 + nloc;
    float* dst = is_f ? fbuf : zbuf;
    const size_t base = (size_t)(c * D + d) * HW + (size_t)h * W + w0;
#pragma unroll
    for (int r = 0; r < 8; ++r) {
        float v = acc[r];
        v = is_f ? (1.0f / (1.0f + __expf(-v))) : fmaxf(v, 0.0f);
        dst[base + (r + mhalf)] = v;
    }
}

// ---------------------------------------------------------- band scan (h -> bf16)
__global__ void scan_qrnn(const float* __restrict__ zbuf, const float* __restrict__ fbuf,
                          const int* __restrict__ revp, int invert,
                          __bf16* __restrict__ hout) {
    int idx = blockIdx.x * blockDim.x + threadIdx.x;
    if (idx >= OCH * HW) return;
    int c = idx / HW, hw = idx % HW;
    bool back = ((revp[0] != 0) != (invert != 0));
    float hacc = 0.0f;
    for (int t = 0; t < D; ++t) {
        int d = back ? (D - 1 - t) : t;
        size_t base = (size_t)(c * D + d) * HW + hw;
        if (t + 1 < D) {
            int dn = back ? (D - 2 - t) : (t + 1);
            __builtin_prefetch(&fbuf[(size_t)(c * D + dn) * HW + hw], 0, 1);
        }
        float f = fbuf[base], z = zbuf[base];
        hacc = f * hacc + (1.0f - f) * z;
        hout[base] = (__bf16)hacc;
    }
}

// ----------------------------------- band scan + 1x1x1 residual conv + bias + add
__global__ void scan_qrnn_res(const float* __restrict__ zbuf, const float* __restrict__ fbuf,
                              const float* __restrict__ x,  const float* __restrict__ Wr,
                              const float* __restrict__ br, const int* __restrict__ revp,
                              int invert, float* __restrict__ out) {
    int idx = blockIdx.x * blockDim.x + threadIdx.x;
    if (idx >= OCH * HW) return;
    int c = idx / HW, hw = idx % HW;
    bool back = ((revp[0] != 0) != (invert != 0));
    float wr[ICH1];
#pragma unroll
    for (int ci = 0; ci < ICH1; ++ci) wr[ci] = Wr[c * ICH1 + ci];
    const float bias = br[c];
    float hacc = 0.0f;
    for (int t = 0; t < D; ++t) {
        int d = back ? (D - 1 - t) : t;
        size_t base = (size_t)(c * D + d) * HW + hw;
        float f = fbuf[base], z = zbuf[base];
        hacc = f * hacc + (1.0f - f) * z;
        float res = bias;
#pragma unroll
        for (int ci = 0; ci < ICH1; ++ci)
            res += wr[ci] * x[(size_t)(ci * D + d) * HW + hw];
        out[base] = hacc + res;
    }
}

extern "C" void kernel_launch(void* const* d_in, const int* in_sizes, int n_in,
                              void* d_out, int out_size, void* d_ws, size_t ws_size,
                              hipStream_t stream) {
    const float* x   = (const float*)d_in[0];
    const float* W1  = (const float*)d_in[1];
    const float* W2  = (const float*)d_in[2];
    const float* Wr  = (const float*)d_in[3];
    const float* br  = (const float*)d_in[4];
    const int*   rev = (const int*)d_in[5];
    float* out = (float*)d_out;

    char* ws = (char*)d_ws;
    size_t off = 0;
    auto alloc = [&](size_t bytes) -> char* {
        char* p = ws + off;
        off = (off + bytes + 255) & ~(size_t)255;
        return p;
    };

    const size_t nx = (size_t)ICH1 * DHW;   // 18,284,544 input elems
    const size_t ng = (size_t)OCH * DHW;    // 36,569,088 per-gate elems

    __bf16* xb  = (__bf16*)alloc(nx * 2);                 // bf16 input
    __bf16* w1p = (__bf16*)alloc((size_t)4 * 14 * 512 * 2);
    __bf16* w2p = (__bf16*)alloc((size_t)4 * 27 * 512 * 2);
    float*  zb  = (float*)alloc(ng * 4);                  // relu gates (reused)
    float*  fb  = (float*)alloc(ng * 4);                  // sigmoid gates (reused)
    __bf16* h1b = (__bf16*)alloc(ng * 2);                 // intermediate h (bf16)

    cvt_f32_bf16<<<(int)((nx + 255) / 256), 256, 0, stream>>>(x, xb, (int)nx);
    pack_w<16, 14><<<112, 256, 0, stream>>>(W1, w1p);     // K=432 -> 14 steps
    pack_w<32, 27><<<216, 256, 0, stream>>>(W2, w2p);     // K=864 -> 27 steps (exact)

    const int nblk = D * H * (W / 16);                    // 71,424 tiles
    conv_qrnn_wmma<16, 14><<<nblk, 128, 0, stream>>>(xb, w1p, zb, fb);
    scan_qrnn<<<(OCH * HW + 255) / 256, 256, 0, stream>>>(zb, fb, rev, 0, h1b);
    conv_qrnn_wmma<32, 27><<<nblk, 128, 0, stream>>>(h1b, w2p, zb, fb);
    scan_qrnn_res<<<(OCH * HW + 255) / 256, 256, 0, stream>>>(zb, fb, x, Wr, br, rev, 1, out);
}
